// GatedIndRnnCellOld_13915694039466
// MI455X (gfx1250) — matrix operand order repaired and checked by reading, not compiled, not run on hardware
//
#include <hip/hip_runtime.h>

// ---------------------------------------------------------------------------
// GatedIndRnnCell fused forward for MI455X (gfx1250, wave32, WMMA bf16)
//   B=8192, IN=2048, H=2048, C=IN+H=4096
//   Double-buffered LDS pipeline, native bf16 conversion, NT output stores.
// ---------------------------------------------------------------------------

#define B_DIM  8192
#define IN_DIM 2048
#define H_DIM  2048
#define C_DIM  (IN_DIM + H_DIM)

typedef __attribute__((ext_vector_type(16))) __bf16        v16bf;
typedef __attribute__((ext_vector_type(2)))  __bf16        v2bf;
typedef __attribute__((ext_vector_type(8)))  float         v8f;
typedef __attribute__((ext_vector_type(4)))  unsigned int  v4u;
typedef __attribute__((ext_vector_type(4)))  float         v4f;

union Frag {
    v16bf f;
    v4u   q[2];
};

#if __has_builtin(__builtin_amdgcn_cvt_pk_bf16_f32)
__device__ __forceinline__ unsigned int pack_bf16(float a, float b) {
    union { v2bf v; unsigned int u; } c;
    c.v = __builtin_amdgcn_cvt_pk_bf16_f32(a, b);   // v_cvt_pk_bf16_f32 (RNE)
    return c.u;
}
#else
// Native fptrunc: let the gfx1250 backend pick the hardware bf16 convert.
__device__ __forceinline__ unsigned int pack_bf16(float a, float b) {
    union { v2bf v; unsigned int u; } c;
    c.v.x = (__bf16)a;
    c.v.y = (__bf16)b;
    return c.u;
}
#endif

// Thread t owns row t/2, half-row (16 floats) t%2 of a 128x32 tile.
__device__ __forceinline__ void gload16(const float* __restrict__ src, int stride,
                                        int tid, v4f r[4]) {
    const v4f* p = (const v4f*)(src + (size_t)(tid >> 1) * stride + (tid & 1) * 16);
    r[0] = p[0]; r[1] = p[1]; r[2] = p[2]; r[3] = p[3];
}

__device__ __forceinline__ void sstore16(const v4f r[4],
                                         unsigned short* __restrict__ lds, int tid) {
    v4u q0, q1;
    q0.x = pack_bf16(r[0].x, r[0].y); q0.y = pack_bf16(r[0].z, r[0].w);
    q0.z = pack_bf16(r[1].x, r[1].y); q0.w = pack_bf16(r[1].z, r[1].w);
    q1.x = pack_bf16(r[2].x, r[2].y); q1.y = pack_bf16(r[2].z, r[2].w);
    q1.z = pack_bf16(r[3].x, r[3].y); q1.w = pack_bf16(r[3].z, r[3].w);
    v4u* dst = (v4u*)&lds[(tid >> 1) * 32 + (tid & 1) * 16];
    dst[0] = q0;
    dst[1] = q1;
}

// A fragment (16-bit 16x32): lane m, lo half-wave K 0..15,
// hi half-wave K 16..31 (matches B fragment). LDS rows are 64B.
__device__ __forceinline__ void load_afrag(const unsigned short* __restrict__ buf,
                                           int row, int hi, Frag& a) {
    const v4u* ap = (const v4u*)&buf[row * 32];
    a.q[0] = ap[hi * 2];
    a.q[1] = ap[hi * 2 + 1];
}

// B fragment (16-bit 32x16): lane n holds 16 contiguous K values
// (lo half-wave K 0..15, hi half-wave K 16..31) = weight-row bytes.
__device__ __forceinline__ void load_bfrag(const unsigned short* __restrict__ buf,
                                           int row, int hi, Frag& b) {
    const v4u* bp = (const v4u*)&buf[row * 32];
    b.q[0] = bp[hi * 2];
    b.q[1] = bp[hi * 2 + 1];
}

__global__ __launch_bounds__(256) void gated_indrnn_wmma(
    const float* __restrict__ input,   // [B, IN]
    const float* __restrict__ state,   // [B, H]
    const float* __restrict__ gw,      // [H, C]
    const float* __restrict__ gbias,   // [H]
    const float* __restrict__ wi,      // [H, IN]
    float* __restrict__ out)           // concatenated outputs
{
    __shared__ alignas(16) unsigned short As[2][128 * 32];  // activations (bf16)
    __shared__ alignas(16) unsigned short Gs[2][128 * 32];  // gate weights (bf16)
    __shared__ alignas(16) unsigned short Ws[2][128 * 32];  // value weights (bf16)

    const int tid  = threadIdx.x;
    const int wave = tid >> 5;
    const int lane = tid & 31;
    const int wm   = wave & 3;    // wave M position: rows wm*32 .. +32
    const int wn   = wave >> 2;   // wave N position: cols wn*64 .. +64
    const int lr   = lane & 15;
    const int hi   = lane >> 4;   // half-wave selector

    const int hn0 = blockIdx.x * 128;   // H (output col) tile base
    const int bm0 = blockIdx.y * 128;   // B (output row) tile base

    v8f acc_g[2][4];   // pre-gate accumulators (f32)
    v8f acc_v[2][4];   // values accumulators (f32)
    #pragma unroll
    for (int i = 0; i < 2; ++i)
        #pragma unroll
        for (int j = 0; j < 4; ++j) {
            v8f z = {};
            acc_g[i][j] = z;
            acc_v[i][j] = z;
        }

    // ======== Phase 1: K over IN (A = input, both weight matrices) =========
    {
        v4f ra[4], rg[4], rw[4];
        gload16(input + (size_t)bm0 * IN_DIM, IN_DIM, tid, ra);
        gload16(gw    + (size_t)hn0 * C_DIM,  C_DIM,  tid, rg);
        gload16(wi    + (size_t)hn0 * IN_DIM, IN_DIM, tid, rw);
        sstore16(ra, As[0], tid);
        sstore16(rg, Gs[0], tid);
        sstore16(rw, Ws[0], tid);
        __syncthreads();

        for (int kt = 0; kt < IN_DIM / 32; ++kt) {
            const int  cur  = kt & 1;
            const int  nxt  = cur ^ 1;
            const bool more = (kt + 1 < IN_DIM / 32);
            if (more) {   // issue next-K global loads before the WMMA block
                const int k = (kt + 1) * 32;
                gload16(input + (size_t)bm0 * IN_DIM + k, IN_DIM, tid, ra);
                gload16(gw    + (size_t)hn0 * C_DIM  + k, C_DIM,  tid, rg);
                gload16(wi    + (size_t)hn0 * IN_DIM + k, IN_DIM, tid, rw);
            }

            Frag a[2], bg[4], bi[4];
            #pragma unroll
            for (int i = 0; i < 2; ++i)
                load_afrag(As[cur], wm * 32 + i * 16 + lr, hi, a[i]);
            #pragma unroll
            for (int j = 0; j < 4; ++j) {
                load_bfrag(Gs[cur], wn * 64 + j * 16 + lr, hi, bg[j]);
                load_bfrag(Ws[cur], wn * 64 + j * 16 + lr, hi, bi[j]);
            }
            #pragma unroll
            for (int i = 0; i < 2; ++i)
                #pragma unroll
                for (int j = 0; j < 4; ++j) {
                    acc_g[i][j] = __builtin_amdgcn_wmma_f32_16x16x32_bf16(
                        false, a[i].f, false, bg[j].f, (short)0, acc_g[i][j], false, false);
                    acc_v[i][j] = __builtin_amdgcn_wmma_f32_16x16x32_bf16(
                        false, a[i].f, false, bi[j].f, (short)0, acc_v[i][j], false, false);
                }

            if (more) {   // convert + stage into alternate buffer during WMMA
                sstore16(ra, As[nxt], tid);
                sstore16(rg, Gs[nxt], tid);
                sstore16(rw, Ws[nxt], tid);
            }
            __syncthreads();
        }
    }

    // ======== Phase 2: K over H (A = state, gate weight cols IN..C) ========
    {
        v4f ra[4], rg[4];
        gload16(state + (size_t)bm0 * H_DIM,          H_DIM, tid, ra);
        gload16(gw    + (size_t)hn0 * C_DIM + IN_DIM, C_DIM, tid, rg);
        sstore16(ra, As[0], tid);
        sstore16(rg, Gs[0], tid);
        __syncthreads();

        for (int kt = 0; kt < H_DIM / 32; ++kt) {
            const int  cur  = kt & 1;
            const int  nxt  = cur ^ 1;
            const bool more = (kt + 1 < H_DIM / 32);
            if (more) {
                const int k = (kt + 1) * 32;
                gload16(state + (size_t)bm0 * H_DIM + k,          H_DIM, tid, ra);
                gload16(gw    + (size_t)hn0 * C_DIM + IN_DIM + k, C_DIM, tid, rg);
            }

            Frag a[2], bg[4];
            #pragma unroll
            for (int i = 0; i < 2; ++i)
                load_afrag(As[cur], wm * 32 + i * 16 + lr, hi, a[i]);
            #pragma unroll
            for (int j = 0; j < 4; ++j)
                load_bfrag(Gs[cur], wn * 64 + j * 16 + lr, hi, bg[j]);
            #pragma unroll
            for (int i = 0; i < 2; ++i)
                #pragma unroll
                for (int j = 0; j < 4; ++j)
                    acc_g[i][j] = __builtin_amdgcn_wmma_f32_16x16x32_bf16(
                        false, a[i].f, false, bg[j].f, (short)0, acc_g[i][j], false, false);

            if (more) {
                sstore16(ra, As[nxt], tid);
                sstore16(rg, Gs[nxt], tid);
            }
            __syncthreads();
        }
    }

    // ======== Fused epilogue ===============================================
    // d_out layout (floats): new_h | concat | pre_gate | gate | values | pre_h
    const size_t BH = (size_t)B_DIM * H_DIM;
    const size_t BC = (size_t)B_DIM * C_DIM;
    float* out_newh = out;
    float* out_pg   = out + BH + BC;
    float* out_gate = out + 2 * BH + BC;
    float* out_val  = out + 3 * BH + BC;
    float* out_ph   = out + 4 * BH + BC;

    // C/D layout (ISA 7.12.2): VGPR r holds M = r + 8*hi, lane&15 = N.
    #pragma unroll
    for (int j = 0; j < 4; ++j) {
        const int h = hn0 + wn * 64 + j * 16 + lr;
        const float bias = gbias[h];
        #pragma unroll
        for (int i = 0; i < 2; ++i) {
            const int row_base = bm0 + wm * 32 + i * 16 + hi * 8;
            #pragma unroll
            for (int r = 0; r < 8; ++r) {
                const size_t idx = (size_t)(row_base + r) * H_DIM + h;
                const float pg = acc_g[i][j][r] + bias;
                const float g  = fminf(fmaxf(pg, 0.0f), 1.0f);   // hard sigmoid
                const float v  = tanhf(acc_v[i][j][r]);
                const float st = state[idx];
                const float ph = st * (1.0f - g) + v * g;
                const float nh = fmaxf(ph, 0.0f);                // relu
                __builtin_nontemporal_store(nh, &out_newh[idx]);
                __builtin_nontemporal_store(pg, &out_pg[idx]);
                __builtin_nontemporal_store(g,  &out_gate[idx]);
                __builtin_nontemporal_store(v,  &out_val[idx]);
                __builtin_nontemporal_store(ph, &out_ph[idx]);
            }
        }
    }
}

// Pure bandwidth kernel: concat_input = [input | state], rows of 4096 floats.
__global__ __launch_bounds__(256) void concat_copy(
    const float* __restrict__ input, const float* __restrict__ state,
    float* __restrict__ out_concat)
{
    const size_t idx = (size_t)blockIdx.x * blockDim.x + threadIdx.x; // float4 idx
    const size_t row = idx >> 10;        // 1024 float4 per concat row
    const size_t c4  = idx & 1023;
    const v4f* src = (c4 < 512)
        ? ((const v4f*)input + row * 512 + c4)
        : ((const v4f*)state + row * 512 + (c4 - 512));
    __builtin_nontemporal_store(*src, (v4f*)out_concat + row * 1024 + c4);
}

extern "C" void kernel_launch(void* const* d_in, const int* in_sizes, int n_in,
                              void* d_out, int out_size, void* d_ws, size_t ws_size,
                              hipStream_t stream) {
    const float* input = (const float*)d_in[0];
    const float* state = (const float*)d_in[1];
    const float* gw    = (const float*)d_in[2];
    const float* gbias = (const float*)d_in[3];
    const float* wi    = (const float*)d_in[4];
    float* out = (float*)d_out;

    dim3 grid(H_DIM / 128, B_DIM / 128);   // 16 x 64 tiles
    gated_indrnn_wmma<<<grid, 256, 0, stream>>>(input, state, gw, gbias, wi, out);

    const size_t BH = (size_t)B_DIM * H_DIM;
    const size_t n4 = (size_t)B_DIM * C_DIM / 4;     // float4 count
    concat_copy<<<(unsigned)((n4 + 255) / 256), 256, 0, stream>>>(
        input, state, out + BH);
}
